// BasicLayer_14705968022124
// MI455X (gfx1250) — matrix operand order repaired and checked by reading, not compiled, not run on hardware
//
#include <hip/hip_runtime.h>
#include <hip/hip_bf16.h>

typedef __attribute__((ext_vector_type(16))) __bf16        v16bf;
typedef __attribute__((ext_vector_type(8)))  float         v8f;
typedef __attribute__((ext_vector_type(8)))  unsigned int  v8u;

#define HW 16384          // 128*128
#define NB 8              // batch
#define NCH 128           // output channels per conv

// round-to-nearest-even fp32 -> bf16 bit pattern (used only in weight swizzle)
__device__ __forceinline__ unsigned short f2bf(float f) {
  unsigned u = __builtin_bit_cast(unsigned, f);
  unsigned r = u + 0x7FFFu + ((u >> 16) & 1u);
  return (unsigned short)(r >> 16);
}

// ---------------------------------------------------------------------------
// Pre-swizzle OIHW fp32 weights into exact per-lane WMMA B fragments (bf16).
// K ordering: k = tap*C + c  (tap = ky*3+kx). Fragment storage:
//   dst[ ((ntile*Ksteps + kstep)*32 + lane)*16 + j ]
// B 32x16 bf16 layout: lanes 0-15 hold K=0..7,16..23 (VGPR v: K=2v,2v+1, +8 for v>=4),
//                      lanes 16-31 hold K=8..15,24..31 (same, +8).
// ---------------------------------------------------------------------------
__global__ void swz_w_kernel(const float* __restrict__ W,
                             unsigned short* __restrict__ dst,
                             int C, int Ksteps) {
  int e = blockIdx.x * blockDim.x + threadIdx.x;
  int total = Ksteps * 512 * 8;            // = K * 128
  if (e >= total) return;
  int ntile = e / (Ksteps * 512);
  int rem   = e % (Ksteps * 512);
  int kstep = rem / 512;
  int r2    = rem % 512;
  int lane  = r2 / 16;
  int j     = r2 % 16;
  int v = j >> 1, hb = j & 1;
  int kl = ((lane >> 4) ? 8 : 0) + ((v >> 2) << 4) + ((v & 3) << 1) + hb;   // B layout K index
  int k  = kstep * 32 + kl;
  int n  = ntile * 16 + (lane & 15);
  int tap = k / C;
  int c   = k - tap * C;
  float f = W[((size_t)n * C + c) * 9 + tap];        // OIHW
  dst[e] = f2bf(f);
}

// ---------------------------------------------------------------------------
// Implicit-GEMM 3x3 SAME conv, bf16 WMMA, f32 accumulate, bias + leaky-ReLU.
// Block = 256 threads = 8 waves; each wave owns a 16-pixel x 128-channel strip
// (8 accumulator tiles; one A fragment per K-step reused 8x).
// Templated on C so all B-fragment offsets (nt*Ksteps*1024 B) and A offsets
// (kl*HW*4 B) fold into global_load immediates and clause.
// SAME padding: out-of-image lanes redirect their *base pointer* to a 2 MB
// zero page once per K-step (2 cndmasks) instead of per-element selects, so
// all 16 A gathers are unconditional immediate-offset loads in one clause.
// B is preloaded in two 4-tile groups to cap register pressure: group-1 loads
// overlap the WMMAs that consume group-0.
// ---------------------------------------------------------------------------
template<int C>
__global__ __launch_bounds__(256)
void conv3x3_wmma_kernel(const float* __restrict__ pa,
                         const float* __restrict__ pbadj,
                         const unsigned short* __restrict__ wfrag,
                         const float* __restrict__ bias,
                         const float* __restrict__ zbuf,
                         float* __restrict__ out) {
  constexpr int Ksteps = (C / 32) * 9;
  constexpr size_t Kbytes = (size_t)Ksteps * 32 * 16 * 2;  // bytes per N-tile of frags

  __shared__ float lds[8][NCH][16];

  const int tid  = threadIdx.x;
  const int wv   = tid >> 5;
  const int lane = tid & 31;
  const int half = lane >> 4;
  const int nl   = lane & 15;

  const int gs = blockIdx.x * 8 + wv;      // global strip id
  const int b  = gs >> 10;                 // 1024 strips per batch image
  const int s  = gs & 1023;
  const int y  = s >> 3;
  const int x0 = (s & 7) << 4;

  v8f acc[8];
#pragma unroll
  for (int nt = 0; nt < 8; ++nt)
#pragma unroll
    for (int i = 0; i < 8; ++i) acc[nt][i] = 0.0f;

  const size_t ib = (size_t)b * 128 * HW;  // batch offset (channel-combined space)
  const int koff = half ? 8 : 0;           // A-layout K offset for upper lane half

  // running B-fragment base (advances 1024 B per K-step)
  const char* bwf = (const char*)wfrag + (size_t)lane * 32;

  for (int tap = 0; tap < 9; ++tap) {
    const int dy = tap / 3 - 1, dx = tap % 3 - 1;
    const int yy = y + dy;
    const int xx = x0 + nl + dx;           // A row M = nl -> pixel x
    const bool ok = ((unsigned)yy < 128u) & ((unsigned)xx < 128u);
    const int sp = yy * 128 + xx;          // deref'd only when ok
    const float* pa2 = pa    + ib + sp;
    const float* pb2 = pbadj + ib + sp;

    for (int cb = 0; cb < C; cb += 32, bwf += 1024) {
      // ---- B fragments, group 0 (tiles 0..3): one clause ----
      v8u bu0[4];
#pragma unroll
      for (int t = 0; t < 4; ++t)
        bu0[t] = *(const v8u*)(bwf + (size_t)t * Kbytes);

      // ---- A gathers: base-pointer select to zero page, then one clause ----
      const float* apr = ((C > 128) && (cb >= 128)) ? pb2 : pa2;
      const float* ap  = ok ? (apr + (size_t)cb * HW) : zbuf;   // 2 cndmasks
      float fv[16];
#pragma unroll
      for (int v = 0; v < 8; ++v)
#pragma unroll
        for (int hb = 0; hb < 2; ++hb) {
          const int kl = ((v < 4) ? 0 : 16) + ((v & 3) << 1) + hb;
          fv[v * 2 + hb] = ap[(size_t)(kl + koff) * HW];
        }

      // ---- native bf16 convert (v_cvt_pk_bf16_f32), frees fv ----
      v16bf af;
#pragma unroll
      for (int j = 0; j < 16; ++j) af[j] = (__bf16)fv[j];

      // ---- B fragments, group 1 (tiles 4..7): overlap WMMAs below ----
      v8u bu1[4];
#pragma unroll
      for (int t = 0; t < 4; ++t)
        bu1[t] = *(const v8u*)(bwf + (size_t)(4 + t) * Kbytes);

      // ---- 8 WMMAs, accumulators persistent in VGPRs ----
#pragma unroll
      for (int t = 0; t < 4; ++t) {
        const v16bf bf = __builtin_bit_cast(v16bf, bu0[t]);
        acc[t] = __builtin_amdgcn_wmma_f32_16x16x32_bf16(
            false, af, false, bf, (short)0, acc[t], false, false);
      }
#pragma unroll
      for (int t = 0; t < 4; ++t) {
        const v16bf bf = __builtin_bit_cast(v16bf, bu1[t]);
        acc[4 + t] = __builtin_amdgcn_wmma_f32_16x16x32_bf16(
            false, af, false, bf, (short)0, acc[4 + t], false, false);
      }
    }
  }

  // ---- epilogue: bias + leaky-ReLU, LDS transpose, coalesced NCHW store ----
#pragma unroll
  for (int nt = 0; nt < 8; ++nt) {
    const int n = nt * 16 + nl;            // C/D layout: lanes hold N, VGPR r holds M
    const float bn = bias[n];
#pragma unroll
    for (int r = 0; r < 8; ++r) {
      const int m = half ? (r + 8) : r;
      float vv = acc[nt][r] + bn;
      vv = vv > 0.0f ? vv : 0.01f * vv;
      lds[wv][n][m] = vv;
    }
  }
  __syncthreads();
  const size_t ob = (size_t)b * NCH * HW + (size_t)y * 128 + x0;
  for (int i = 0; i < 64; ++i) {
    const int ch = i * 2 + half;
    out[ob + (size_t)ch * HW + nl] = lds[wv][ch][nl];
  }
}

// ---------------------------------------------------------------------------
// Fused tail: ds = 1x1 conv (2 ch), 2-class entropy map, _unob stencil.
// One thread per pixel; channel loop is coalesced across x.
// ---------------------------------------------------------------------------
__global__ __launch_bounds__(256)
void ds_amap_kernel(const float* __restrict__ h2,
                    const float* __restrict__ wds,
                    const int* __restrict__ tar,
                    float* __restrict__ dsout,
                    float* __restrict__ amapout) {
  int p = blockIdx.x * blockDim.x + threadIdx.x;
  if (p >= NB * HW) return;
  const int b = p >> 14;
  const int r = p & (HW - 1);
  const int y = r >> 7;
  const int x = r & 127;

  const size_t hb = (size_t)b * NCH * HW + r;
  float d0 = 0.0f, d1 = 0.0f;
  for (int c = 0; c < NCH; ++c) {
    const float hv = h2[hb + (size_t)c * HW];
    d0 += hv * wds[c];
    d1 += hv * wds[NCH + c];
  }
  dsout[(size_t)b * 2 * HW + r]      = d0;
  dsout[(size_t)b * 2 * HW + HW + r] = d1;

  // entropy of softmax([d0,d1]) / log2(2)
  const float m  = fmaxf(d0, d1);
  const float e0 = __expf(d0 - m), e1 = __expf(d1 - m);
  const float inv = 1.0f / (e0 + e1);
  const float p0 = e0 * inv, p1 = e1 * inv;
  const float ent = -(p0 * __log2f(p0 + 1e-6f) + p1 * __log2f(p1 + 1e-6f));
  float amap = (ent >= 0.001f) ? 1.0f : 0.0f;

  const int tb = b * HW;
  if (y > 0 && y < 127 && x > 0 && x < 127) {
    const int c  = tar[tb + r];
    const int nb = tar[tb + r + 128] + tar[tb + r - 128] +
                   tar[tb + r + 1]   + tar[tb + r - 1];
    if ((c != 0) && (4 * c != nb)) amap = 0.0f;
  }
  amapout[p] = amap;
}

// ---------------------------------------------------------------------------
extern "C" void kernel_launch(void* const* d_in, const int* in_sizes, int n_in,
                              void* d_out, int out_size, void* d_ws, size_t ws_size,
                              hipStream_t stream) {
  const float* x    = (const float*)d_in[0];
  const float* skip = (const float*)d_in[1];
  const int*   tar  = (const int*)  d_in[2];
  const float* W1   = (const float*)d_in[3];
  const float* b1   = (const float*)d_in[4];
  const float* W2   = (const float*)d_in[5];
  const float* b2   = (const float*)d_in[6];
  const float* Wds  = (const float*)d_in[7];

  const size_t H2ELEMS = (size_t)NB * NCH * HW;   // 16,777,216
  const size_t DSELEMS = (size_t)NB * 2 * HW;     // 262,144

  // workspace: h1 fp32 (64 MB) | wfrag1 (576 KB) | wfrag2 (288 KB) | zeros (2 MB)
  float*          h1  = (float*)d_ws;
  unsigned short* wf1 = (unsigned short*)((char*)d_ws + H2ELEMS * sizeof(float));
  unsigned short* wf2 = wf1 + 294912;
  float*          zb  = (float*)((char*)(wf2 + 147456));

  float* h2out   = (float*)d_out;
  float* dsout   = h2out + H2ELEMS;
  float* amapout = dsout + DSELEMS;

  const int K1steps = (256 / 32) * 9;   // 72
  const int K2steps = (128 / 32) * 9;   // 36

  // 0) zero page for SAME-padding reads (re-cleared every launch)
  hipMemsetAsync(zb, 0, (size_t)32 * HW * sizeof(float), stream);

  // 1) swizzle weights into WMMA B fragments
  swz_w_kernel<<<(294912 + 255) / 256, 256, 0, stream>>>(W1, wf1, 256, K1steps);
  swz_w_kernel<<<(147456 + 255) / 256, 256, 0, stream>>>(W2, wf2, 128, K2steps);

  // 2) conv1: concat(x,skip) -> h1 (ws, fp32)
  conv3x3_wmma_kernel<256><<<1024, 256, 0, stream>>>(
      x, skip - (size_t)128 * HW, wf1, b1, zb, h1);

  // 3) conv2: h1 -> h2 (d_out head)
  conv3x3_wmma_kernel<128><<<1024, 256, 0, stream>>>(
      h1, h1, wf2, b2, zb, h2out);

  // 4) ds + entropy map + unob
  ds_amap_kernel<<<(NB * HW + 255) / 256, 256, 0, stream>>>(
      h2out, Wds, tar, dsout, amapout);
}
